// VectorQuantizerEMAWithIndices_32323923869927
// MI455X (gfx1250) — matrix operand order repaired and checked by reading, not compiled, not run on hardware
//
#include <hip/hip_runtime.h>
#include <math.h>

typedef __attribute__((ext_vector_type(16))) _Float16 v16h;
typedef __attribute__((ext_vector_type(8)))  float    v8f;
typedef __attribute__((ext_vector_type(4)))  unsigned int u32x4;
typedef __attribute__((ext_vector_type(8)))  int      i32x8;
typedef __attribute__((ext_vector_type(4)))  int      i32x4;

#define CDIM   256
#define KCODE  1024
#define NVEC   32768          // 32*32*32
#define MT     2048           // NVEC/16 row tiles
#define UT     64             // KCODE/16 col tiles
#define CCH    8              // CDIM/32 k-chunks
#define ZTOT   8388608        // 32*256*32*32
#define TILE_B 8192           // bytes per staged B tile (8 frags * 32 lanes * 16 halves * 2B)

// ---- workspace byte offsets ----
#define WS_ZA   0ull                                   // f16 swizzled z fragments  (16 MB)
#define WS_EB   (WS_ZA + (size_t)NVEC*CDIM*2)          // f16 swizzled codebook     (512 KB)
#define WS_ZN   (WS_EB + (size_t)KCODE*CDIM*2)         // fp32 ||z||^2 [N]
#define WS_EN   (WS_ZN + (size_t)NVEC*4)               // fp32 ||e||^2 [K]
#define WS_IDX  (WS_EN + (size_t)KCODE*4)              // int32 argmin [N]
#define WS_CNT  (WS_IDX + (size_t)NVEC*4)              // fp32 counts [K]
#define WS_DW   (WS_CNT + (size_t)KCODE*4)             // fp32 dw [K,C]
#define WS_SCL  (WS_DW + (size_t)KCODE*CDIM*4)         // scalars: [0]=lossAcc [1]=nsum

// ---- output float offsets (concatenated tuple, reference return order) ----
#define OUT_Q    0
#define OUT_LOSS 8388608
#define OUT_PERP 8388609
#define OUT_TOK  8388610
#define OUT_NCS  8421378
#define OUT_EMAW 8422402
#define OUT_EMB  8684546

// -------------------- zero accumulators (counts + dw + scalars) --------------------
__global__ void vq_zero(float* acc, int n) {
    int i = blockIdx.x * blockDim.x + threadIdx.x;
    if (i < n) acc[i] = 0.0f;
}

// -------- swizzle codebook -> WMMA B fragments (B[k][n]: lane=n(+khalf), halves=k) --------
__global__ void vq_prep_eB(const float* __restrict__ emb, _Float16* __restrict__ eB) {
    int tid  = blockIdx.x * blockDim.x + threadIdx.x;   // UT*CCH*32 = 16384
    int u    = tid >> 8;
    int c    = (tid >> 5) & 7;
    int lane = tid & 31;
    int n    = lane & 15;
    int kbase = c * 32 + ((lane >= 16) ? 8 : 0);
    const float* src = emb + (size_t)(u * 16 + n) * CDIM + kbase;
    v16h frag;
#pragma unroll
    for (int h = 0; h < 16; ++h) frag[h] = (_Float16)src[(h < 8) ? h : (h + 8)];
    *(v16h*)(eB + ((size_t)(u * CCH + c) * 32 + lane) * 16) = frag;
}

__global__ void vq_enorm(const float* __restrict__ emb, float* __restrict__ en) {
    int k = blockIdx.x * blockDim.x + threadIdx.x;      // KCODE
    const float4* r = (const float4*)(emb + (size_t)k * CDIM);
    float s = 0.0f;
#pragma unroll 4
    for (int i = 0; i < CDIM / 4; ++i) { float4 v = r[i]; s += v.x*v.x + v.y*v.y + v.z*v.z + v.w*v.w; }
    en[k] = s;
}

// -------- swizzle z_e [B,C,H,W] -> WMMA A fragments (A[m][k]: lane=m(+khalf), halves=k) --------
__global__ void vq_prep_zA(const float* __restrict__ z, _Float16* __restrict__ zA) {
    int tid  = blockIdx.x * blockDim.x + threadIdx.x;   // MT*CCH*32 = 524288
    int t    = tid >> 8;
    int c    = (tid >> 5) & 7;
    int lane = tid & 31;
    int g    = lane >> 4;
    int mg   = t * 16 + (lane & 15);
    int b    = mg >> 10;
    int rem  = mg & 1023;
    const float* base = z + (size_t)b * (CDIM * 1024) + rem;
    v16h frag;
#pragma unroll
    for (int h = 0; h < 16; ++h) {
        int k = c * 32 + ((h < 8) ? ((g ? 8 : 0) + h) : ((g ? 24 : 16) + (h - 8)));
        frag[h] = (_Float16)base[(size_t)k * 1024];
    }
    *(v16h*)(zA + ((size_t)(t * CCH + c) * 32 + lane) * 16) = frag;
}

__global__ void vq_znorm(const float* __restrict__ z, float* __restrict__ zn) {
    int n = blockIdx.x * blockDim.x + threadIdx.x;      // NVEC
    int b = n >> 10, rem = n & 1023;
    const float* base = z + (size_t)b * (CDIM * 1024) + rem;
    float s = 0.0f;
#pragma unroll 8
    for (int c = 0; c < CDIM; ++c) { float v = base[(size_t)c * 1024]; s += v * v; }
    zn[n] = s;
}

// -------------------- fused WMMA GEMM + argmin over K --------------------
// 256 threads = 8 waves; each wave owns one 16-row M-tile. Codebook tiles are
// DMA'd into LDS by the Tensor Data Mover (1KB slice per wave), double-buffered
// on TENSORcnt, and shared by all 8 waves (8x less L2 traffic than per-wave loads).

#if __has_builtin(__builtin_amdgcn_tensor_load_to_lds)
#define VQ_USE_TDM 1
#else
#define VQ_USE_TDM 0
#endif

#if VQ_USE_TDM
// 1-D TDM copy: 128 elements x 8B = 1KB, global -> LDS.
// amdgpu-toolchain (clang-23) 6-arg form:
//   (u32x4 g0, i32x8 g1, i32x4 g2, i32x4 g3, i32x8 extra, i32 cpol)
__device__ __forceinline__ void tdm_issue_1k(const _Float16* gsrc, unsigned ldsOff) {
    unsigned long long ga = (unsigned long long)(size_t)gsrc;
    u32x4 g0;
    g0[0] = 1u;                                         // count=1, user descriptor
    g0[1] = ldsOff;                                     // lds_addr (bytes)
    g0[2] = (unsigned)ga;                               // global_addr[31:0]
    g0[3] = (unsigned)((ga >> 32) & 0x01FFFFFFu) | (2u << 30);  // addr[56:32] | type=2
    i32x8 g1;
    g1[0] = 0x30000;        // workgroup_mask=0, data_size=3 (8B)
    g1[1] = 128 << 16;      // tensor_dim0=128 (low 16 bits in [31:16])
    g1[2] = 1 << 16;        // tensor_dim0 hi=0 | tensor_dim1=1
    g1[3] = 128 << 16;      // tensor_dim1 hi=0 | tile_dim0=128
    g1[4] = 1;              // tile_dim1=1, tile_dim2=0
    g1[5] = 128;            // tensor_dim0_stride lo
    g1[6] = 0;
    g1[7] = 0;
    i32x4 z4 = {0, 0, 0, 0};
    i32x8 z8 = {0, 0, 0, 0, 0, 0, 0, 0};
    __builtin_amdgcn_tensor_load_to_lds(g0, g1, z4, z4, z8, 0);
}
#endif

__global__ void __launch_bounds__(256)
vq_gemm_argmin(const _Float16* __restrict__ zA, const _Float16* __restrict__ eB,
               const float* __restrict__ zn, const float* __restrict__ en,
               int* __restrict__ idxOut) {
    __shared__ _Float16 sb[2][TILE_B / 2];              // 2 x 8KB double buffer
    int wave = threadIdx.x >> 5;
    int lane = threadIdx.x & 31;
    int tile = blockIdx.x * 8 + wave;                   // [0, MT)
    int ncol = lane & 15;
    int mbase = tile * 16 + ((lane >= 16) ? 8 : 0);

    const _Float16* aBase = zA + (size_t)tile * CCH * 32 * 16;
    v16h a[CCH];
#pragma unroll
    for (int c = 0; c < CCH; ++c)
        a[c] = *(const v16h*)(aBase + ((size_t)c * 32 + lane) * 16);

    float znOwn[8];
#pragma unroll
    for (int v = 0; v < 8; ++v) znOwn[v] = zn[mbase + v];

    float best[8]; int bidx[8];
#pragma unroll
    for (int v = 0; v < 8; ++v) { best[v] = 3.4e38f; bidx[v] = 0; }

#if VQ_USE_TDM
    unsigned ldsBase = (unsigned)(size_t)&sb[0][0];     // low 32 bits = LDS byte offset
    // prologue: each wave DMAs its 1KB slice of tile 0 into buffer 0
    tdm_issue_1k(eB + (size_t)wave * 512, ldsBase + wave * 1024);

    for (int u = 0; u < UT - 1; ++u) {
        int buf = u & 1;
        // kick DMA of next tile into the other buffer (prev reads of it are
        // fenced by the trailing barrier of iteration u-1)
        tdm_issue_1k(eB + (size_t)(u + 1) * (TILE_B / 2) + wave * 512,
                     ldsBase + (buf ^ 1) * TILE_B + wave * 1024);
        __builtin_amdgcn_s_wait_tensorcnt(1);           // tile u's slice landed
        __syncthreads();                                 // all 8 slices landed
        const _Float16* bs = &sb[buf][0];
        v8f acc = {};
#pragma unroll
        for (int c = 0; c < CCH; ++c) {
            v16h bf = *(const v16h*)(bs + (c * 32 + lane) * 16);
            acc = __builtin_amdgcn_wmma_f32_16x16x32_f16(
                false, a[c], false, bf, (short)0, acc, false, false);
        }
        int   col = u * 16 + ncol;
        float ecn = en[col];
#pragma unroll
        for (int v = 0; v < 8; ++v) {
            float d = znOwn[v] + ecn - 2.0f * acc[v];
            if (d < best[v]) { best[v] = d; bidx[v] = col; }
        }
        __syncthreads();                                 // done reading buf
    }
    // peeled last tile (constant wait immediates)
    {
        __builtin_amdgcn_s_wait_tensorcnt(0);
        __syncthreads();
        const _Float16* bs = &sb[(UT - 1) & 1][0];
        v8f acc = {};
#pragma unroll
        for (int c = 0; c < CCH; ++c) {
            v16h bf = *(const v16h*)(bs + (c * 32 + lane) * 16);
            acc = __builtin_amdgcn_wmma_f32_16x16x32_f16(
                false, a[c], false, bf, (short)0, acc, false, false);
        }
        int   col = (UT - 1) * 16 + ncol;
        float ecn = en[col];
#pragma unroll
        for (int v = 0; v < 8; ++v) {
            float d = znOwn[v] + ecn - 2.0f * acc[v];
            if (d < best[v]) { best[v] = d; bidx[v] = col; }
        }
    }
#else
    // fallback: register double-buffered global loads of B fragments
    v16h bcur[CCH], bnxt[CCH];
#pragma unroll
    for (int c = 0; c < CCH; ++c)
        bcur[c] = *(const v16h*)(eB + ((size_t)c * 32 + lane) * 16);
    for (int u = 0; u < UT; ++u) {
        if (u + 1 < UT) {
            const _Float16* nb = eB + (size_t)(u + 1) * CCH * 32 * 16;
#pragma unroll
            for (int c = 0; c < CCH; ++c)
                bnxt[c] = *(const v16h*)(nb + ((size_t)c * 32 + lane) * 16);
        }
        v8f acc = {};
#pragma unroll
        for (int c = 0; c < CCH; ++c)
            acc = __builtin_amdgcn_wmma_f32_16x16x32_f16(
                false, a[c], false, bcur[c], (short)0, acc, false, false);
        int   col = u * 16 + ncol;
        float ecn = en[col];
#pragma unroll
        for (int v = 0; v < 8; ++v) {
            float d = znOwn[v] + ecn - 2.0f * acc[v];
            if (d < best[v]) { best[v] = d; bidx[v] = col; }
        }
#pragma unroll
        for (int c = 0; c < CCH; ++c) bcur[c] = bnxt[c];
    }
#endif

    // cross-lane argmin within each 16-lane half (row m lives in one half-group)
#pragma unroll
    for (int off = 1; off < 16; off <<= 1) {
#pragma unroll
        for (int v = 0; v < 8; ++v) {
            float od = __shfl_xor(best[v], off, 32);
            int   oi = __shfl_xor(bidx[v], off, 32);
            if (od < best[v] || (od == best[v] && oi < bidx[v])) { best[v] = od; bidx[v] = oi; }
        }
    }
    if (ncol == 0) {
#pragma unroll
        for (int v = 0; v < 8; ++v) idxOut[mbase + v] = bidx[v];
    }
}

// -------------------- counts (segment_sum of ones) + token_map --------------------
__global__ void vq_counts(const int* __restrict__ idx, float* __restrict__ counts,
                          float* __restrict__ tok) {
    int n = blockIdx.x * blockDim.x + threadIdx.x;      // NVEC
    int k = idx[n];
    atomicAdd(&counts[k], 1.0f);
    tok[n] = (float)k;
}

// ------- fused: q_out (straight-through value == quantized), loss partial, dw scatter -------
__global__ void vq_quant(const float* __restrict__ z, const float* __restrict__ emb,
                         const int* __restrict__ idx, float* __restrict__ qOut,
                         float* __restrict__ dw, float* __restrict__ lossAcc) {
    __shared__ float red[256];
    int tid = blockIdx.x * blockDim.x + threadIdx.x;    // ZTOT
    int rem = tid & 1023;
    int c   = (tid >> 10) & 255;
    int b   = tid >> 18;
    int n   = b * 1024 + rem;
    int k   = idx[n];
    float zv = z[tid];
    float qv = emb[(size_t)k * CDIM + c];
    qOut[tid] = qv;                                     // z + stopgrad(q - z) == q
    float d = qv - zv;
    atomicAdd(&dw[(size_t)k * CDIM + c], zv);
    red[threadIdx.x] = d * d;
    __syncthreads();
    for (int s = 128; s > 0; s >>= 1) {
        if (threadIdx.x < s) red[threadIdx.x] += red[threadIdx.x + s];
        __syncthreads();
    }
    if (threadIdx.x == 0) atomicAdd(lossAcc, red[0]);
}

// -------- single-block finalize of small stats: ncs, n, perplexity, loss --------
__global__ void vq_stats(const float* __restrict__ counts, const float* __restrict__ ecs,
                         float* __restrict__ out, float* __restrict__ scl) {
    __shared__ float sN[1024];
    __shared__ float sP[1024];
    int k = threadIdx.x;                                // 1024 threads
    float cnt = counts[k];
    float ncs = ecs[k] * 0.99f + 0.01f * cnt;
    out[OUT_NCS + k] = ncs;
    float p = cnt / (float)NVEC;
    sN[k] = ncs;
    sP[k] = p * logf(p + 1e-10f);
    __syncthreads();
    for (int s = 512; s > 0; s >>= 1) {
        if (k < s) { sN[k] += sN[k + s]; sP[k] += sP[k + s]; }
        __syncthreads();
    }
    if (k == 0) {
        scl[1] = sN[0];
        out[OUT_PERP] = expf(-sP[0]);
        out[OUT_LOSS] = 0.25f * scl[0] / (float)ZTOT;
    }
}

// -------------------- EMA weight update + normalized embedding --------------------
__global__ void vq_finalize(const float* __restrict__ emaw, const float* __restrict__ dw,
                            const float* __restrict__ scl, float* __restrict__ out) {
    int tid = blockIdx.x * blockDim.x + threadIdx.x;    // KCODE*CDIM
    int k = tid >> 8;
    float w = emaw[tid] * 0.99f + 0.01f * dw[tid];
    out[OUT_EMAW + tid] = w;
    float nsum = scl[1];
    float ncs  = out[OUT_NCS + k];
    float cs   = (ncs + 1e-10f) / (nsum + (float)KCODE * 1e-10f) * nsum;
    out[OUT_EMB + tid] = w / cs;
}

extern "C" void kernel_launch(void* const* d_in, const int* in_sizes, int n_in,
                              void* d_out, int out_size, void* d_ws, size_t ws_size,
                              hipStream_t stream) {
    const float* z_e  = (const float*)d_in[0];
    const float* emb  = (const float*)d_in[1];
    const float* ecs  = (const float*)d_in[2];
    const float* emaw = (const float*)d_in[3];
    float* out = (float*)d_out;

    char* ws = (char*)d_ws;
    _Float16* zA  = (_Float16*)(ws + WS_ZA);
    _Float16* eB  = (_Float16*)(ws + WS_EB);
    float* zn     = (float*)(ws + WS_ZN);
    float* en     = (float*)(ws + WS_EN);
    int*   idx    = (int*)  (ws + WS_IDX);
    float* counts = (float*)(ws + WS_CNT);
    float* dw     = (float*)(ws + WS_DW);
    float* scl    = (float*)(ws + WS_SCL);

    int nAcc = KCODE + KCODE * CDIM + 8;
    vq_zero<<<(nAcc + 255) / 256, 256, 0, stream>>>(counts, nAcc);
    vq_prep_eB<<<UT * CCH * 32 / 256, 256, 0, stream>>>(emb, eB);
    vq_enorm<<<KCODE / 256, 256, 0, stream>>>(emb, en);
    vq_prep_zA<<<MT * CCH * 32 / 256, 256, 0, stream>>>(z_e, zA);
    vq_znorm<<<NVEC / 256, 256, 0, stream>>>(z_e, zn);
    vq_gemm_argmin<<<MT / 8, 256, 0, stream>>>(zA, eB, zn, en, idx);
    vq_counts<<<NVEC / 256, 256, 0, stream>>>(idx, counts, out + OUT_TOK);
    vq_quant<<<ZTOT / 256, 256, 0, stream>>>(z_e, emb, idx, out + OUT_Q, dw, &scl[0]);
    vq_stats<<<1, 1024, 0, stream>>>(counts, ecs, out, scl);
    vq_finalize<<<KCODE * CDIM / 256, 256, 0, stream>>>(emaw, dw, scl, out);
}
